// AM_Softmax_88553635709481
// MI455X (gfx1250) — compile-verified, matrix operand order checked by reading, not run on hardware
//
#include <hip/hip_runtime.h>

// AM-Softmax loss for MI455X (gfx1250): wave32, WMMA f16 16x16x32, TDM
// (tensor_load_to_lds) double-buffered B-tile staging with native LDS padding.
//
// Workspace layout (bytes):
//   wN   : 100000*128 f16 = 25,600,000
//   fN   :   2048*128 f16 =    524,288
//   psum : 32*2048 f32    =    262,144   (per-slab sum of 2^((x-30)*log2e))
//   csum : 32*2048 f32    =    262,144   (per-slab sum of clamped cos)
// total ~26.6 MB.

typedef _Float16 v4h  __attribute__((ext_vector_type(4)));
typedef _Float16 v8h  __attribute__((ext_vector_type(8)));
typedef _Float16 v16h __attribute__((ext_vector_type(16)));
typedef float    v8f  __attribute__((ext_vector_type(8)));
typedef unsigned int u32x4 __attribute__((ext_vector_type(4)));
typedef int          i32x8 __attribute__((ext_vector_type(8)));
typedef int          i32x4 __attribute__((ext_vector_type(4)));

#define BN 2048
#define DK 128
#define CC 100000
#define MARGIN 0.3f
#define SCALE  30.0f
#define EPS_   0.1f
#define LOG2E  1.44269504088896340736f
#define LN2    0.69314718055994530942f

#define MT 128            // rows per workgroup tile
#define NT 64             // logit-columns per tile iteration
#define NSLAB 32          // column slabs (grid.x)
#define NTILES ((CC + NT - 1) / NT)   // 1563
#define LDT 136           // padded LDS row stride in halves (272B = 68 banks)

// ---------------------------------------------------------------------------
// Row L2-normalize f32 -> f16. One wave per row (8 rows / 256-thread block).
// ---------------------------------------------------------------------------
__global__ __launch_bounds__(256) void normalize_rows_f16(
    const float* __restrict__ src, _Float16* __restrict__ dst, int nrows)
{
  const int wave = threadIdx.x >> 5;
  const int lane = threadIdx.x & 31;
  const int row  = blockIdx.x * 8 + wave;
  if (row >= nrows) return;

  const float4 v = *(const float4*)(src + (size_t)row * DK + lane * 4);
  float ss = v.x * v.x + v.y * v.y + v.z * v.z + v.w * v.w;
#pragma unroll
  for (int off = 16; off >= 1; off >>= 1) ss += __shfl_xor(ss, off, 32);
  const float rn = rsqrtf(ss);

  v4h h;
  h[0] = (_Float16)(v.x * rn);
  h[1] = (_Float16)(v.y * rn);
  h[2] = (_Float16)(v.z * rn);
  h[3] = (_Float16)(v.w * rn);
  *(v4h*)(dst + (size_t)row * DK + lane * 4) = h;
}

// ---------------------------------------------------------------------------
// TDM: DMA one B tile (NT weight rows x 256B) into LDS at lds_byte_addr with
// 16B padding after every 256B row (-> padded stride LDT halves).
// 2D descriptor, data_size = 8B units. Rows past (CC - cbase) read as zero.
// ---------------------------------------------------------------------------
__device__ __forceinline__ void tdm_issue_b_tile(
    const _Float16* __restrict__ wN, int cbase, unsigned lds_byte_addr)
{
  const unsigned long long ga =
      (unsigned long long)(uintptr_t)(wN + (size_t)cbase * DK);
  const unsigned td1 = (unsigned)(CC - cbase);   // tensor_dim1 (rows avail)

  u32x4 g0;
  g0[0] = 1u;                                    // count=1 (valid), user mode
  g0[1] = lds_byte_addr;                         // lds_addr
  g0[2] = (unsigned)(ga & 0xFFFFFFFFull);        // global_addr[31:0]
  g0[3] = (unsigned)(ga >> 32) | (2u << 30);     // global_addr[56:32] | type=2

  i32x8 g1;
  // data_size=3 (8B) | pad_enable | pad_interval=5 (64 DW) | pad_amount=3 (4 DW)
  g1[0] = (int)((3u << 16) | (1u << 20) | (5u << 22) | (3u << 25));
  g1[1] = (int)(32u << 16);                      // tensor_dim0 = 32 units
  g1[2] = (int)((td1 & 0xFFFFu) << 16);          // dim0 hi=0 | tensor_dim1 lo
  g1[3] = (int)(((td1 >> 16) & 0xFFFFu) | (32u << 16)); // dim1 hi | tile_dim0=32
  g1[4] = (int)NT;                               // tile_dim1=64 | tile_dim2=0
  g1[5] = 32;                                    // tensor_dim0_stride = 32 units
  g1[6] = 0;                                     // stride0 hi | stride1 lo
  g1[7] = 0;                                     // stride1 hi

  const i32x4 gz4 = {0, 0, 0, 0};                // groups 2/3 unused (2D)
  const i32x8 gz8 = {0, 0, 0, 0, 0, 0, 0, 0};    // trailing group (zero-filled)
  __builtin_amdgcn_tensor_load_to_lds(g0, g1, gz4, gz4, gz8, 0);
}

// Load one 16x32 f16 WMMA B fragment from a padded LDS tile.
__device__ __forceinline__ v16h load_bfrag(const _Float16* __restrict__ bbase,
                                           int brow, int k0)
{
  v8h lo = *(const v8h*)&bbase[brow * LDT + k0];
  v8h hi = *(const v8h*)&bbase[brow * LDT + k0 + 16];
  return __builtin_shufflevector(lo, hi, 0, 1, 2, 3, 4, 5, 6, 7,
                                 8, 9, 10, 11, 12, 13, 14, 15);
}

// ---------------------------------------------------------------------------
// Fused cosine-GEMM + streaming softmax partials.
// grid = (NSLAB, BN/MT), block = 256 (8 waves). Wave w owns rows w*16..w*16+15.
// A tile staged once, fragments hoisted to VGPRs; its LDS footprint is then
// reused as two TDM-filled B buffers (double buffered). B fragments are
// software-pipelined one step deep so LDS latency hides behind each WMMA.
// ---------------------------------------------------------------------------
__global__ __launch_bounds__(256) void amsoftmax_gemm(
    const _Float16* __restrict__ fN, const _Float16* __restrict__ wN,
    float* __restrict__ psum, float* __restrict__ csum)
{
  __shared__ __attribute__((aligned(16))) _Float16 smem[MT * LDT];  // 34816 B

  const int tid  = threadIdx.x;
  const int lane = tid & 31;
  const int wave = tid >> 5;
  const int slab = blockIdx.x;
  const int rblk = blockIdx.y;

  // Stage the A tile (MT x DK halves) once: 2048 chunks of 8 halves.
  for (int ch = tid; ch < MT * (DK / 8); ch += 256) {
    const int r  = ch >> 4;
    const int kc = (ch & 15) * 8;
    *(v8h*)&smem[r * LDT + kc] =
        *(const v8h*)(fN + (size_t)(rblk * MT + r) * DK + kc);
  }

  const int arow = wave * 16 + (lane & 15);
  const int kb   = (lane >> 4) * 8;   // K sub-offset per WMMA A/B layout

  float ps[8], cs[8];
#pragma unroll
  for (int v = 0; v < 8; ++v) { ps[v] = 0.f; cs[v] = 0.f; }

  __syncthreads();

  // Hoist the 4 A fragments for this wave's 16-row strip (constant all tiles).
  v16h afrag[4];
#pragma unroll
  for (int ks = 0; ks < 4; ++ks) {
    const int k0 = kb + ks * 32;
    v8h lo = *(const v8h*)&smem[arow * LDT + k0];
    v8h hi = *(const v8h*)&smem[arow * LDT + k0 + 16];
    afrag[ks] = __builtin_shufflevector(lo, hi, 0, 1, 2, 3, 4, 5, 6, 7,
                                        8, 9, 10, 11, 12, 13, 14, 15);
  }
  __syncthreads();   // A reads done: smem is now reusable as B double-buffer

  // LDS byte address of the buffers (generic ptr truncates to LDS offset).
  const unsigned ldsBase = (unsigned)(uintptr_t)(void*)&smem[0];
  const unsigned bufB0 = ldsBase;
  const unsigned bufB1 = ldsBase + (unsigned)(NT * LDT * 2);

  if (wave == 0) tdm_issue_b_tile(wN, slab * NT, bufB0);

  int cur = 0;
  for (int t = slab; t < NTILES; t += NSLAB) {
    const int cbase = t * NT;
    const int tn    = t + NSLAB;
    __syncthreads();                 // all readers of buf[cur^1] are done
    if (wave == 0) {
      if (tn < NTILES) {
        tdm_issue_b_tile(wN, tn * NT, cur ? bufB0 : bufB1);
        __builtin_amdgcn_s_wait_tensorcnt(1);   // oldest (= buf[cur]) done
      } else {
        __builtin_amdgcn_s_wait_tensorcnt(0);
      }
    }
    __syncthreads();                 // buf[cur] contents visible to all waves

    const _Float16* bbase = smem + (cur ? NT * LDT : 0);

    v8f acc[4] = {};
    v16h bf_cur = load_bfrag(bbase, lane & 15, kb);   // ks=0, n=0
#pragma unroll
    for (int ks = 0; ks < 4; ++ks) {
      const int k0 = kb + ks * 32;
#pragma unroll
      for (int n = 0; n < 4; ++n) {
        v16h bf_next;
        if (!(ks == 3 && n == 3)) {
          const int nn  = (n == 3) ? 0 : (n + 1);
          const int kn  = (n == 3) ? (k0 + 32) : k0;
          bf_next = load_bfrag(bbase, nn * 16 + (lane & 15), kn);
        }
        acc[n] = __builtin_amdgcn_wmma_f32_16x16x32_f16(
            false, afrag[ks], false, bf_cur, (short)0, acc[n], false, false);
        bf_cur = bf_next;
      }
    }

    // Streaming softmax epilogue (fixed max bound: logits <= SCALE).
    const bool tail = (cbase + NT > CC);
#pragma unroll
    for (int n = 0; n < 4; ++n) {
      const int c = cbase + n * 16 + (lane & 15);
      const float keep = (!tail || (c < CC)) ? 1.0f : 0.0f;  // branchless
#pragma unroll
      for (int v = 0; v < 8; ++v) {
        const float cv = __builtin_amdgcn_fmed3f(acc[n][v], -1.f, 1.f);
        const float u  = __builtin_fmaf(cv, SCALE * LOG2E, -SCALE * LOG2E);
        const float p  = __builtin_amdgcn_exp2f(u);   // u in [-86.6, 0]: safe
        ps[v] = __builtin_fmaf(keep, p, ps[v]);
        cs[v] = __builtin_fmaf(keep, cv, cs[v]);
      }
    }
    cur ^= 1;
  }

  // Combine the 16 lanes that share each output row (xor within 16-lane group).
#pragma unroll
  for (int v = 0; v < 8; ++v) {
#pragma unroll
    for (int off = 8; off >= 1; off >>= 1) {
      ps[v] += __shfl_xor(ps[v], off, 32);
      cs[v] += __shfl_xor(cs[v], off, 32);
    }
  }
  if ((lane & 15) == 0) {
    // lanes 0..15 hold rows strip+0..7 ; lanes 16..31 hold rows strip+8..15
    const int rg = rblk * MT + wave * 16 + (lane >> 4) * 8;
#pragma unroll
    for (int v = 0; v < 8; ++v) {
      psum[slab * BN + rg + v] = ps[v];
      csum[slab * BN + rg + v] = cs[v];
    }
  }
}

// ---------------------------------------------------------------------------
// Combine slab partials, apply the margin fix-up at the label column, compute
// label-smoothed CE, reduce to the scalar mean loss. Single block.
// ---------------------------------------------------------------------------
__global__ __launch_bounds__(256) void amsoftmax_finalize(
    const _Float16* __restrict__ fN, const _Float16* __restrict__ wN,
    const int* __restrict__ labels,
    const float* __restrict__ psum, const float* __restrict__ csum,
    float* __restrict__ out)
{
  __shared__ float red[256];
  const int tid = threadIdx.x;
  float acc = 0.f;

  for (int r = tid; r < BN; r += 256) {
    float st = 0.f, ct = 0.f;
    for (int s = 0; s < NSLAB; ++s) {
      st += psum[s * BN + r];
      ct += csum[s * BN + r];
    }
    const int lbl = labels[r];
    const _Float16* fr = fN + (size_t)r * DK;
    const _Float16* wr = wN + (size_t)lbl * DK;
    float dot = 0.f;
#pragma unroll 8
    for (int k = 0; k < DK; ++k) dot += (float)fr[k] * (float)wr[k];

    const float cosy = __builtin_amdgcn_fmed3f(dot, -1.f, 1.f);
    const float xy_plain = SCALE * cosy;            // logit without margin
    const float xy       = SCALE * (cosy - MARGIN); // logit with margin
    const float p_old = __builtin_amdgcn_exp2f((xy_plain - SCALE) * LOG2E);
    const float p_new = __builtin_amdgcn_exp2f((xy - SCALE) * LOG2E);
    st = st - p_old + p_new;                         // swap label term
    const float sum_logits = SCALE * ct - SCALE * MARGIN;

    const float logZ  = SCALE + __builtin_amdgcn_logf(st) * LN2;
    const float nll   = logZ - xy;
    const float smooth = logZ - sum_logits * (1.0f / CC);
    acc += (1.f - EPS_) * nll + EPS_ * smooth;
  }

  red[tid] = acc;
  __syncthreads();
  for (int off = 128; off >= 1; off >>= 1) {
    if (tid < off) red[tid] += red[tid + off];
    __syncthreads();
  }
  if (tid == 0) out[0] = red[0] * (1.f / BN);
}

// ---------------------------------------------------------------------------
extern "C" void kernel_launch(void* const* d_in, const int* in_sizes, int n_in,
                              void* d_out, int out_size, void* d_ws,
                              size_t ws_size, hipStream_t stream)
{
  (void)in_sizes; (void)n_in; (void)out_size; (void)ws_size;
  const float* features = (const float*)d_in[0];
  const int*   labels   = (const int*)d_in[1];
  const float* weight   = (const float*)d_in[2];
  float* out = (float*)d_out;

  _Float16* wN = (_Float16*)d_ws;
  _Float16* fN = wN + (size_t)CC * DK;
  float* psum  = (float*)(fN + (size_t)BN * DK);
  float* csum  = psum + (size_t)NSLAB * BN;

  normalize_rows_f16<<<dim3((CC + 7) / 8), 256, 0, stream>>>(weight, wN, CC);
  normalize_rows_f16<<<dim3((BN + 7) / 8), 256, 0, stream>>>(features, fN, BN);
  amsoftmax_gemm<<<dim3(NSLAB, BN / MT), 256, 0, stream>>>(fN, wN, psum, csum);
  amsoftmax_finalize<<<1, 256, 0, stream>>>(fN, wN, labels, psum, csum, out);
}
